// VisionEncoderMambaBlock_5317169512745
// MI455X (gfx1250) — matrix operand order repaired and checked by reading, compiled, not run on hardware
//
#include <hip/hip_runtime.h>
#include <hip/hip_bf16.h>

// ---------------- problem constants ----------------
#define DIM      768
#define DT_RANK  48
#define D_STATE  16
#define BATCH    4
#define SEQ      1024
#define MROWS    (BATCH * SEQ)            // 4096
#define DBC_N    (DT_RANK + 2 * D_STATE)  // 80
#define DT_KPAD  64                       // DT_RANK padded to multiple of 32
#define EPSF     1e-5f
#define CH       16                       // scan chunks
#define CLEN     (SEQ / CH)               // 64 steps per chunk

typedef __bf16 bf16;
typedef __attribute__((address_space(3))) __bf16 lds_bf16;
typedef __attribute__((ext_vector_type(16))) __bf16 v16bf;
typedef __attribute__((ext_vector_type(8)))  float  v8f;

__device__ __forceinline__ float softplus_f(float x) {
    return (x > 20.f) ? x : log1pf(__expf(x));
}
__device__ __forceinline__ float silu_f(float x) {
    return x / (1.f + __expf(-x));
}

// ---------------- CDNA5 async copy global -> LDS ----------------
__device__ __forceinline__ void async_copy_b128(lds_bf16* lp, const bf16* gp) {
    asm volatile("global_load_async_to_lds_b128 %0, %1, off"
                 :: "v"(lp), "v"(gp)
                 : "memory");
}
__device__ __forceinline__ void wait_async(int more_in_flight) {
    if (more_in_flight) {
        asm volatile("s_wait_asynccnt 6" ::: "memory");   // next stage's 6 copies stay in flight
    } else {
        asm volatile("s_wait_asynccnt 0" ::: "memory");
    }
}

// ---------------- WMMA fragment loaders (global, direct) ----------------
__device__ __forceinline__ v16bf load_a_frag(const bf16* __restrict__ base,
                                             int ld, int k0, int lane) {
    int m  = lane & 15;
    int hi = lane >> 4;
    const bf16* p = base + (size_t)m * ld + k0 + hi * 8;
    union { v16bf v; uint4 q[2]; } f;
    f.q[0] = *(const uint4*)(p);
    f.q[1] = *(const uint4*)(p + 16);
    return f.v;
}
__device__ __forceinline__ v16bf load_b_frag(const bf16* __restrict__ wbase,
                                             int ld, int k0, int lane) {
    int n  = lane & 15;
    int hi = lane >> 4;
    const bf16* p = wbase + (size_t)n * ld + k0 + hi * 8;
    union { v16bf v; uint4 q[2]; } f;
    f.q[0] = *(const uint4*)(p);
    f.q[1] = *(const uint4*)(p + 16);
    return f.v;
}

// ---------------- WMMA fragment loaders (LDS, padded rows) ----------------
#define ROWP 72   // 144B row stride: 16B-aligned, conflict-free b128 reads
__device__ __forceinline__ v16bf lds_a_frag(const bf16* base, int kk, int lane) {
    int m  = lane & 15;
    int hi = lane >> 4;
    const bf16* p = base + m * ROWP + kk + hi * 8;
    union { v16bf v; uint4 q[2]; } f;
    f.q[0] = *(const uint4*)(p);
    f.q[1] = *(const uint4*)(p + 16);
    return f.v;
}
__device__ __forceinline__ v16bf lds_b_frag(const bf16* base, int kk, int lane) {
    int n  = lane & 15;
    int hi = lane >> 4;
    const bf16* p = base + n * ROWP + kk + hi * 8;
    union { v16bf v; uint4 q[2]; } f;
    f.q[0] = *(const uint4*)(p);
    f.q[1] = *(const uint4*)(p + 16);
    return f.v;
}

// ================= big GEMM: LDS-staged, async double-buffered =================
// Block: 256 thr = 8 waves (4 M x 2 N), tile 64(M) x 128(N), K-stage 64.
template <int MODE>   // 0: linear -> f32+bf16 ; 1: softplus -> f32+bf16
__global__ __launch_bounds__(256)
void gemm_wmma_lds_kernel(const bf16* __restrict__ A,
                          const bf16* __restrict__ W,
                          const float* __restrict__ bias, int K,
                          float* __restrict__ outF, bf16* __restrict__ outH) {
    __shared__ __align__(16) bf16 shA[2][64 * ROWP];
    __shared__ __align__(16) bf16 shB[2][128 * ROWP];

    const int tid  = threadIdx.x;
    const int wave = tid >> 5;
    const int lane = tid & 31;
    const int wr   = wave & 3;
    const int wc   = wave >> 2;
    const int mBase = blockIdx.y * 64;
    const int nBase = blockIdx.x * 128;

    auto issue_stage = [&](int k0, int buf) {
#pragma unroll
        for (int i = 0; i < 2; ++i) {
            int c = tid + i * 256;
            int row = c >> 3, c8 = c & 7;
            async_copy_b128((lds_bf16*)(&shA[buf][0] + row * ROWP + c8 * 8),
                            A + (size_t)(mBase + row) * DIM + k0 + c8 * 8);
        }
#pragma unroll
        for (int i = 0; i < 4; ++i) {
            int c = tid + i * 256;
            int row = c >> 3, c8 = c & 7;
            async_copy_b128((lds_bf16*)(&shB[buf][0] + row * ROWP + c8 * 8),
                            W + (size_t)(nBase + row) * DIM + k0 + c8 * 8);
        }
    };

    v8f acc[4];
#pragma unroll
    for (int i = 0; i < 4; ++i)
        acc[i] = (v8f){0.f, 0.f, 0.f, 0.f, 0.f, 0.f, 0.f, 0.f};

    const int NS = K / 64;
    issue_stage(0, 0);
    issue_stage(64, 1);

    for (int ks = 0; ks < NS; ++ks) {
        const int buf = ks & 1;
        wait_async(ks + 1 < NS);
        __syncthreads();

        // hoist ALL fragment loads for this stage, then issue 8 back-to-back WMMAs
        const bf16* aT = &shA[buf][wr * 16 * ROWP];
        v16bf afr[2];
        v16bf bfr[2][4];
#pragma unroll
        for (int c = 0; c < 2; ++c) {
            afr[c] = lds_a_frag(aT, c * 32, lane);
#pragma unroll
            for (int i = 0; i < 4; ++i)
                bfr[c][i] = lds_b_frag(&shB[buf][(wc * 64 + i * 16) * ROWP], c * 32, lane);
        }
#pragma unroll
        for (int c = 0; c < 2; ++c)
#pragma unroll
            for (int i = 0; i < 4; ++i)
                acc[i] = __builtin_amdgcn_wmma_f32_16x16x32_bf16(
                    false, afr[c], false, bfr[c][i], (short)0, acc[i], false, false);

        __syncthreads();
        if (ks + 2 < NS) issue_stage((ks + 2) * 64, buf);
    }

    const int hi = lane >> 4;
    const int nl = lane & 15;
#pragma unroll
    for (int i = 0; i < 4; ++i) {
        const int n = nBase + wc * 64 + i * 16 + nl;
        const float bv = bias[n];
#pragma unroll
        for (int r = 0; r < 8; ++r) {
            const int m = mBase + wr * 16 + r + 8 * hi;
            float v = acc[i][r] + bv;
            if (MODE == 1) v = softplus_f(v);
            outF[(size_t)m * DIM + n] = v;
            outH[(size_t)m * DIM + n] = (bf16)v;
        }
    }
}

// ================= small GEMMs: direct-register WMMA =================
// MODE 2: dbc scatter ; MODE 3: softplus -> outF only
template <int WROWS, int WCOLS, int NF, int MODE>
__global__ __launch_bounds__(WROWS * WCOLS * 32)
void gemm_wmma_direct_kernel(const bf16* __restrict__ A, int lda,
                             const bf16* __restrict__ W, int ldw,
                             const float* __restrict__ bias, int K,
                             float* __restrict__ outF, int ldo,
                             bf16* __restrict__ dOut, float* __restrict__ bOut,
                             float* __restrict__ cOut) {
    const int tid  = threadIdx.x;
    const int wave = tid >> 5;
    const int lane = tid & 31;
    const int wr   = wave % WROWS;
    const int wc   = wave / WROWS;
    const int mBase = (blockIdx.y * WROWS + wr) * 16;
    const int nBase = blockIdx.x * (WCOLS * NF * 16) + wc * (NF * 16);

    v8f acc[NF];
#pragma unroll
    for (int i = 0; i < NF; ++i)
        acc[i] = (v8f){0.f, 0.f, 0.f, 0.f, 0.f, 0.f, 0.f, 0.f};

    const bf16* aBase = A + (size_t)mBase * lda;
    for (int k0 = 0; k0 < K; k0 += 32) {
        // hoist loads, then back-to-back WMMAs
        v16bf a = load_a_frag(aBase, lda, k0, lane);
        v16bf bfr[NF];
#pragma unroll
        for (int i = 0; i < NF; ++i)
            bfr[i] = load_b_frag(W + (size_t)(nBase + i * 16) * ldw, ldw, k0, lane);
#pragma unroll
        for (int i = 0; i < NF; ++i)
            acc[i] = __builtin_amdgcn_wmma_f32_16x16x32_bf16(
                false, a, false, bfr[i], (short)0, acc[i], false, false);
    }

    const int hi = lane >> 4;
    const int nl = lane & 15;
#pragma unroll
    for (int i = 0; i < NF; ++i) {
        const int n = nBase + i * 16 + nl;
        const float bv = (MODE == 2) ? 0.f : bias[n];
#pragma unroll
        for (int r = 0; r < 8; ++r) {
            const int m = mBase + r + 8 * hi;
            float v = acc[i][r] + bv;
            if (MODE == 3) {
                outF[(size_t)m * ldo + n] = softplus_f(v);
            } else { // MODE 2
                if (n < DT_RANK) {
                    dOut[(size_t)m * DT_KPAD + n] = (bf16)v;
                } else if (n < DT_RANK + D_STATE) {
                    bOut[(size_t)m * D_STATE + (n - DT_RANK)] = v;
                    dOut[(size_t)m * DT_KPAD + n] = (bf16)0.f;
                } else {
                    cOut[(size_t)m * D_STATE + (n - DT_RANK - D_STATE)] = v;
                }
            }
        }
    }
}

// ---------------- LayerNorm -> bf16 ----------------
__global__ __launch_bounds__(256)
void ln_kernel(const float* __restrict__ x, const float* __restrict__ g,
               const float* __restrict__ b, bf16* __restrict__ out) {
    const int row = blockIdx.x;
    const int tid = threadIdx.x;
    const float* xr = x + (size_t)row * DIM;

    float v[3];
    float s = 0.f, s2 = 0.f;
#pragma unroll
    for (int i = 0; i < 3; ++i) {
        v[i] = xr[tid + i * 256];
        s += v[i];
        s2 += v[i] * v[i];
    }
    __shared__ float sh1[256], sh2[256];
    sh1[tid] = s; sh2[tid] = s2;
    __syncthreads();
    for (int off = 128; off > 0; off >>= 1) {
        if (tid < off) { sh1[tid] += sh1[tid + off]; sh2[tid] += sh2[tid + off]; }
        __syncthreads();
    }
    const float mu  = sh1[0] * (1.f / DIM);
    const float var = sh2[0] * (1.f / DIM) - mu * mu;
    const float inv = rsqrtf(var + EPSF);
#pragma unroll
    for (int i = 0; i < 3; ++i) {
        const int c = tid + i * 256;
        out[(size_t)row * DIM + c] = (bf16)((v[i] - mu) * inv * g[c] + b[c]);
    }
}

// ================= chunked selective scan =================
// linear recurrence h' = dA*h + dBu  split into CH chunks of CLEN steps:
//   P1: per-chunk local scan from h=0, accumulating P[n] = prod(dA)
//   P2: cross-chunk state propagation (CH-step serial, per channel)
//   P3: re-run chunk from true initial state, emit y (+ x*D)
__global__ __launch_bounds__(128)
void scan_phase1_kernel(const float* __restrict__ xin,
                        const float* __restrict__ delta,
                        const float* __restrict__ Bm,
                        const float* __restrict__ A_log,
                        float* __restrict__ Pbuf,   // (B, CH, DIM, 16)
                        float* __restrict__ Hbuf) { // (B, CH, DIM, 16)
    const int d = blockIdx.y * 128 + threadIdx.x;
    const int b = blockIdx.x;
    const int c = blockIdx.z;

    float A[D_STATE];
#pragma unroll
    for (int n = 0; n < D_STATE; ++n)
        A[n] = -__expf(A_log[(size_t)d * D_STATE + n]);

    float h[D_STATE], P[D_STATE];
#pragma unroll
    for (int n = 0; n < D_STATE; ++n) { h[n] = 0.f; P[n] = 1.f; }

    const size_t row0 = (size_t)b * SEQ + (size_t)c * CLEN;
    for (int s = 0; s < CLEN; ++s) {
        const size_t row = row0 + s;
        const float dl = delta[row * DIM + d];
        const float xv = xin[row * DIM + d];
        const float du = dl * xv;
        float Bv[D_STATE];
        const float4* Bp = (const float4*)(Bm + row * D_STATE);
#pragma unroll
        for (int j = 0; j < 4; ++j) *(float4*)&Bv[4 * j] = Bp[j];
#pragma unroll
        for (int n = 0; n < D_STATE; ++n) {
            const float dA = __expf(dl * A[n]);
            P[n] *= dA;
            h[n] = dA * h[n] + du * Bv[n];
        }
    }
    const size_t o = (((size_t)b * CH + c) * DIM + d) * D_STATE;
#pragma unroll
    for (int j = 0; j < 4; ++j) {
        *(float4*)&Pbuf[o + 4 * j] = *(float4*)&P[4 * j];
        *(float4*)&Hbuf[o + 4 * j] = *(float4*)&h[4 * j];
    }
}

__global__ __launch_bounds__(128)
void scan_phase2_kernel(const float* __restrict__ Pbuf,
                        const float* __restrict__ Hbuf,
                        float* __restrict__ Hs) {   // (B, CH, DIM, 16) chunk-start states
    const int d = blockIdx.y * 128 + threadIdx.x;
    const int b = blockIdx.x;
    float hs[D_STATE];
#pragma unroll
    for (int n = 0; n < D_STATE; ++n) hs[n] = 0.f;
    for (int c = 0; c < CH; ++c) {
        const size_t o = (((size_t)b * CH + c) * DIM + d) * D_STATE;
#pragma unroll
        for (int j = 0; j < 4; ++j)
            *(float4*)&Hs[o + 4 * j] = *(float4*)&hs[4 * j];
        float P[D_STATE], H[D_STATE];
#pragma unroll
        for (int j = 0; j < 4; ++j) {
            *(float4*)&P[4 * j] = *(const float4*)&Pbuf[o + 4 * j];
            *(float4*)&H[4 * j] = *(const float4*)&Hbuf[o + 4 * j];
        }
#pragma unroll
        for (int n = 0; n < D_STATE; ++n) hs[n] = P[n] * hs[n] + H[n];
    }
}

__global__ __launch_bounds__(128)
void scan_phase3_kernel(const float* __restrict__ xin,
                        const float* __restrict__ delta,
                        const float* __restrict__ Bm,
                        const float* __restrict__ Cm,
                        const float* __restrict__ A_log,
                        const float* __restrict__ Dp,
                        const float* __restrict__ Hs,
                        float* __restrict__ y) {
    const int d = blockIdx.y * 128 + threadIdx.x;
    const int b = blockIdx.x;
    const int c = blockIdx.z;

    float A[D_STATE];
#pragma unroll
    for (int n = 0; n < D_STATE; ++n)
        A[n] = -__expf(A_log[(size_t)d * D_STATE + n]);
    const float Dv = Dp[d];

    float h[D_STATE];
    const size_t o = (((size_t)b * CH + c) * DIM + d) * D_STATE;
#pragma unroll
    for (int j = 0; j < 4; ++j) *(float4*)&h[4 * j] = *(const float4*)&Hs[o + 4 * j];

    const size_t row0 = (size_t)b * SEQ + (size_t)c * CLEN;
    for (int s = 0; s < CLEN; ++s) {
        const size_t row = row0 + s;
        const float dl = delta[row * DIM + d];
        const float xv = xin[row * DIM + d];
        const float du = dl * xv;
        float Bv[D_STATE], Cv[D_STATE];
        const float4* Bp = (const float4*)(Bm + row * D_STATE);
        const float4* Cp = (const float4*)(Cm + row * D_STATE);
#pragma unroll
        for (int j = 0; j < 4; ++j) {
            *(float4*)&Bv[4 * j] = Bp[j];
            *(float4*)&Cv[4 * j] = Cp[j];
        }
        float acc = 0.f;
#pragma unroll
        for (int n = 0; n < D_STATE; ++n) {
            const float dA = __expf(dl * A[n]);
            h[n] = dA * h[n] + du * Bv[n];
            acc += h[n] * Cv[n];
        }
        y[row * DIM + d] = acc + xv * Dv;
    }
}

// ---------------- elementwise combine ----------------
__global__ __launch_bounds__(256)
void combine_kernel(const float* __restrict__ y1, const float* __restrict__ y2,
                    const float* __restrict__ xp, const float* __restrict__ skip,
                    float* __restrict__ out, int n) {
    const int i = blockIdx.x * 256 + threadIdx.x;
    if (i < n) {
        out[i] = (y1[i] + y2[i]) * silu_f(xp[i]) + skip[i];
    }
}

// ---------------- weight conversion ----------------
__global__ __launch_bounds__(256)
void cvt_bf16_kernel(const float* __restrict__ src, bf16* __restrict__ dst, int n) {
    const int i = blockIdx.x * 256 + threadIdx.x;
    if (i < n) dst[i] = (bf16)src[i];
}
__global__ __launch_bounds__(256)
void cvt_wdt_kernel(const float* __restrict__ src, bf16* __restrict__ dst) {
    const int i = blockIdx.x * 256 + threadIdx.x;
    if (i < DIM * DT_KPAD) {
        const int r = i / DT_KPAD, c = i % DT_KPAD;
        dst[i] = (c < DT_RANK) ? (bf16)src[r * DT_RANK + c] : (bf16)0.f;
    }
}

// ---------------- host-side launch ----------------
extern "C" void kernel_launch(void* const* d_in, const int* in_sizes, int n_in,
                              void* d_out, int out_size, void* d_ws, size_t ws_size,
                              hipStream_t stream) {
    const float* x       = (const float*)d_in[0];
    const float* ln_g    = (const float*)d_in[1];
    const float* ln_b    = (const float*)d_in[2];
    const float* W_proj  = (const float*)d_in[3];
    const float* b_proj  = (const float*)d_in[4];
    const float* W_fc    = (const float*)d_in[5];
    const float* b_fc    = (const float*)d_in[6];
    const float* W_bc    = (const float*)d_in[7];
    const float* b_bc    = (const float*)d_in[8];
    const float* W_dbc   = (const float*)d_in[9];
    const float* W_dt    = (const float*)d_in[10];
    const float* b_dt    = (const float*)d_in[11];
    const float* A_log   = (const float*)d_in[12];
    const float* D_param = (const float*)d_in[13];
    float* out = (float*)d_out;

    char* wsp = (char*)d_ws;
    size_t off = 0;
    auto carve = [&](size_t bytes) -> void* {
        void* p = wsp + off;
        off = (off + bytes + 255) & ~(size_t)255;
        return p;
    };
    bf16*  xn_h    = (bf16*) carve((size_t)MROWS * DIM * 2);
    bf16*  Wp_h    = (bf16*) carve((size_t)DIM * DIM * 2);
    bf16*  Wf_h    = (bf16*) carve((size_t)DIM * DIM * 2);
    bf16*  Wb_h    = (bf16*) carve((size_t)DIM * DIM * 2);
    bf16*  Wdbc_h  = (bf16*) carve((size_t)DBC_N * DIM * 2);
    bf16*  Wdt_h   = (bf16*) carve((size_t)DIM * DT_KPAD * 2);
    float* xp_f    = (float*)carve((size_t)MROWS * DIM * 4);
    bf16*  xp_h    = (bf16*) carve((size_t)MROWS * DIM * 2);
    float* xf_f    = (float*)carve((size_t)MROWS * DIM * 4);
    bf16*  xf_h    = (bf16*) carve((size_t)MROWS * DIM * 2);
    float* xb_f    = (float*)carve((size_t)MROWS * DIM * 4);
    bf16*  xb_h    = (bf16*) carve((size_t)MROWS * DIM * 2);
    bf16*  dIn_f   = (bf16*) carve((size_t)MROWS * DT_KPAD * 2);
    bf16*  dIn_b   = (bf16*) carve((size_t)MROWS * DT_KPAD * 2);
    float* Bm_f    = (float*)carve((size_t)MROWS * D_STATE * 4);
    float* Cm_f    = (float*)carve((size_t)MROWS * D_STATE * 4);
    float* Bm_b    = (float*)carve((size_t)MROWS * D_STATE * 4);
    float* Cm_b    = (float*)carve((size_t)MROWS * D_STATE * 4);
    float* delt_f  = (float*)carve((size_t)MROWS * DIM * 4);
    float* delt_b  = (float*)carve((size_t)MROWS * DIM * 4);
    float* y1      = (float*)carve((size_t)MROWS * DIM * 4);
    float* y2      = (float*)carve((size_t)MROWS * DIM * 4);
    float* Pbuf    = (float*)carve((size_t)BATCH * CH * DIM * D_STATE * 4);
    float* Hbuf    = (float*)carve((size_t)BATCH * CH * DIM * D_STATE * 4);
    float* Hsbuf   = (float*)carve((size_t)BATCH * CH * DIM * D_STATE * 4);
    (void)ws_size; (void)in_sizes; (void)n_in; (void)out_size;

    // ---- 1) convert weights to bf16 ----
    {
        int n = DIM * DIM;
        cvt_bf16_kernel<<<(n + 255) / 256, 256, 0, stream>>>(W_proj, Wp_h, n);
        cvt_bf16_kernel<<<(n + 255) / 256, 256, 0, stream>>>(W_fc,   Wf_h, n);
        cvt_bf16_kernel<<<(n + 255) / 256, 256, 0, stream>>>(W_bc,   Wb_h, n);
        int nd = DBC_N * DIM;
        cvt_bf16_kernel<<<(nd + 255) / 256, 256, 0, stream>>>(W_dbc, Wdbc_h, nd);
        int nw = DIM * DT_KPAD;
        cvt_wdt_kernel<<<(nw + 255) / 256, 256, 0, stream>>>(W_dt, Wdt_h);
    }

    // ---- 2) layernorm -> bf16 ----
    ln_kernel<<<MROWS, 256, 0, stream>>>(x, ln_g, ln_b, xn_h);

    // ---- 3) big GEMMs: async-LDS double-buffered WMMA ----
    dim3 gBig(DIM / 128, MROWS / 64);
    gemm_wmma_lds_kernel<0><<<gBig, 256, 0, stream>>>(xn_h, Wp_h, b_proj, DIM, xp_f, xp_h);
    gemm_wmma_lds_kernel<1><<<gBig, 256, 0, stream>>>(xp_h, Wf_h, b_fc,   DIM, xf_f, xf_h);
    gemm_wmma_lds_kernel<1><<<gBig, 256, 0, stream>>>(xp_h, Wb_h, b_bc,   DIM, xb_f, xb_h);

    // ---- 4) dbc GEMMs (N=80) + delta GEMMs (K=64): direct WMMA ----
    dim3 gDbc(1, MROWS / 128);
    gemm_wmma_direct_kernel<8, 1, 5, 2><<<gDbc, 256, 0, stream>>>(
        xf_h, DIM, Wdbc_h, DIM, nullptr, DIM, nullptr, 0, dIn_f, Bm_f, Cm_f);
    gemm_wmma_direct_kernel<8, 1, 5, 2><<<gDbc, 256, 0, stream>>>(
        xb_h, DIM, Wdbc_h, DIM, nullptr, DIM, nullptr, 0, dIn_b, Bm_b, Cm_b);

    gemm_wmma_direct_kernel<4, 2, 4, 3><<<gBig, 256, 0, stream>>>(
        dIn_f, DT_KPAD, Wdt_h, DT_KPAD, b_dt, DT_KPAD, delt_f, DIM,
        nullptr, nullptr, nullptr);
    gemm_wmma_direct_kernel<4, 2, 4, 3><<<gBig, 256, 0, stream>>>(
        dIn_b, DT_KPAD, Wdt_h, DT_KPAD, b_dt, DT_KPAD, delt_b, DIM,
        nullptr, nullptr, nullptr);

    // ---- 5) chunk-parallel selective scans ----
    dim3 gP13(BATCH, DIM / 128, CH);
    dim3 gP2(BATCH, DIM / 128);
    // forward path (xf)
    scan_phase1_kernel<<<gP13, 128, 0, stream>>>(xf_f, delt_f, Bm_f, A_log, Pbuf, Hbuf);
    scan_phase2_kernel<<<gP2, 128, 0, stream>>>(Pbuf, Hbuf, Hsbuf);
    scan_phase3_kernel<<<gP13, 128, 0, stream>>>(xf_f, delt_f, Bm_f, Cm_f, A_log, D_param, Hsbuf, y1);
    // backward path (xb) — buffers reused (stream-ordered)
    scan_phase1_kernel<<<gP13, 128, 0, stream>>>(xb_f, delt_b, Bm_b, A_log, Pbuf, Hbuf);
    scan_phase2_kernel<<<gP2, 128, 0, stream>>>(Pbuf, Hbuf, Hsbuf);
    scan_phase3_kernel<<<gP13, 128, 0, stream>>>(xb_f, delt_b, Bm_b, Cm_b, A_log, D_param, Hsbuf, y2);

    // ---- 6) combine: (y1 + y2) * silu(xp) + skip ----
    {
        int n = MROWS * DIM;
        combine_kernel<<<(n + 255) / 256, 256, 0, stream>>>(y1, y2, xp_f, x, out, n);
    }
}
